// VisionMamba_17111149707730
// MI455X (gfx1250) — matrix operand order repaired and checked by reading, not compiled, not run on hardware
//
#include <hip/hip_runtime.h>
#include <math.h>
#include <stdint.h>

typedef __bf16 bf16;
typedef __bf16 v16bf __attribute__((ext_vector_type(16)));
typedef __bf16 v8bf  __attribute__((ext_vector_type(8)));
typedef float  v8f   __attribute__((ext_vector_type(8)));
typedef unsigned int v4u __attribute__((ext_vector_type(4)));
typedef int v4i __attribute__((ext_vector_type(4)));
typedef int v8i __attribute__((ext_vector_type(8)));

namespace {
constexpr int BATCH  = 32;
constexpr int TOKL   = 196;            // 14x14 patches
constexpr int MROWS  = BATCH * TOKL;   // 6272 (= 49*128)
constexpr int DM     = 768;
constexpr int DEPTH  = 12;
constexpr int DI     = 1536;
constexpr int DXZ    = 2 * DI;         // 3072
constexpr int DTR    = 49;
constexpr int DST    = 16;
constexpr int NXPRAW = DTR + 2 * DST;  // 81
constexpr int NXP    = 128;            // padded x_proj out dim
constexpr int NCLS   = 1000;
constexpr int NH     = 1024;           // padded classes
constexpr int MH     = 128;            // padded batch rows for head GEMM
constexpr int DCONV  = 4;
}

#if defined(__gfx1250__) && __has_builtin(__builtin_amdgcn_tensor_load_to_lds)
#define USE_TDM 1
#else
#define USE_TDM 0
#endif

#if defined(__gfx1250__)
__device__ __forceinline__ void wait_tensorcnt0() {
#if __has_builtin(__builtin_amdgcn_s_wait_tensorcnt)
  __builtin_amdgcn_s_wait_tensorcnt(0);
#else
  asm volatile("s_wait_tensorcnt 0x0" ::: "memory");
#endif
}
#endif

#if USE_TDM
// Issue one 2D TDM tile load: tile (rows x 32 bf16) from a row-major matrix
// with row stride k_elems (elements), into contiguous LDS at lds_addr.
// D# bit layout per CDNA5 ISA (§8.3/8.4): group0 holds
// count/lds_addr/global_addr/type, group1 holds data_size/dims/tile/strides.
__device__ __forceinline__ void tdm_load_tile_2d(unsigned long long gaddr,
                                                 unsigned lds_addr,
                                                 unsigned k_elems, int rows) {
  v4u g0;
  g0[0] = 1u;                                   // count = 1 valid descriptor
  g0[1] = lds_addr;                             // LDS byte address
  g0[2] = (unsigned)gaddr;                      // global_addr[31:0]
  g0[3] = (unsigned)((gaddr >> 32) & 0x01FFFFFFull) | (2u << 30);  // [56:32] | type=2
  const unsigned dim1 = 1u << 20;               // tensor_dim1: large (no clipping)
  v8i g1;
  g1[0] = (int)(1u << 16);                      // data_size = 1 (2 bytes, bf16)
  g1[1] = (int)((k_elems & 0xFFFFu) << 16);     // tensor_dim0[15:0]
  g1[2] = (int)((k_elems >> 16) | ((dim1 & 0xFFFFu) << 16));   // dim0 hi | dim1 lo
  g1[3] = (int)((dim1 >> 16) | (32u << 16));    // dim1 hi | tile_dim0 = 32 (BK)
  g1[4] = (int)(unsigned)rows;                  // tile_dim1 = rows, tile_dim2 = 0
  g1[5] = (int)k_elems;                         // tensor_dim0_stride[31:0]
  g1[6] = 0;                                    // stride hi | dim1_stride lo
  g1[7] = 0;
  const v4i z4 = {0, 0, 0, 0};
#if __clang_major__ >= 23
  const v8i z8 = {0, 0, 0, 0, 0, 0, 0, 0};
  __builtin_amdgcn_tensor_load_to_lds(g0, g1, z4, z4, z8, 0);
#else
  __builtin_amdgcn_tensor_load_to_lds(g0, g1, z4, z4, 0);
#endif
}
#endif

// ---------------------------------------------------------------------------
// bf16 WMMA GEMM:  C[M,N] (f32) = A[M,K](bf16,row-major) * B[N,K](bf16,row-major)^T
// M%128==0, N%128==0, K%32==0.  Optional bias[col] and residual add.
// Block = 256 threads = 8 waves; block tile 128x128; each wave owns a 32x64
// region = 8 x v_wmma_f32_16x16x32_bf16 per 32-wide K chunk.
// K chunks staged in LDS by the Tensor Data Mover (double buffered), with a
// cooperative global->LDS fallback when the TDM builtin is unavailable.
// ---------------------------------------------------------------------------
__launch_bounds__(256)
__global__ void gemm_bf16_wmma(const bf16* __restrict__ A,
                               const bf16* __restrict__ Bw,
                               const float* __restrict__ bias,
                               const float* resid,
                               float* C, int Mm, int Nn, int Kk) {
  (void)Mm;
  constexpr int BM = 128, BN = 128, BK = 32;
  __shared__ bf16 sA[2][BM][BK];
  __shared__ bf16 sB[2][BN][BK];

  const int tid  = threadIdx.x;
  const int lane = tid & 31;
  const int wid  = tid >> 5;
  const int half = lane >> 4;
  const int r    = lane & 15;
  const int wm   = (wid & 3) * 32;   // wave M offset (0..96)
  const int wn   = (wid >> 2) * 64;  // wave N offset (0,64)
  const int m0   = blockIdx.y * BM;
  const int n0   = blockIdx.x * BN;

  const v8f vz = {0.f, 0.f, 0.f, 0.f, 0.f, 0.f, 0.f, 0.f};
  v8f acc[2][4];
#pragma unroll
  for (int mt = 0; mt < 2; ++mt)
#pragma unroll
    for (int nt = 0; nt < 4; ++nt) acc[mt][nt] = vz;

  // one K-chunk of WMMAs from LDS buffer `bufi`
  auto compute_chunk = [&](int bufi) {
    v16bf bfrag[4];
#pragma unroll
    for (int nt = 0; nt < 4; ++nt)
    {
      v8bf blo = *(const v8bf*)&sB[bufi][wn + nt * 16 + r][half << 3];
      v8bf bhi = *(const v8bf*)&sB[bufi][wn + nt * 16 + r][16 + (half << 3)];
#pragma unroll
      for (int i = 0; i < 8; ++i) { bfrag[nt][i] = blo[i]; bfrag[nt][i + 8] = bhi[i]; }
    }
#pragma unroll
    for (int mt = 0; mt < 2; ++mt) {
      // A fragment per ISA 16-bit layout: K chunks [8*half, +8) and [16+8*half, +8)
      v8bf alo = *(const v8bf*)&sA[bufi][wm + mt * 16 + r][half << 3];
      v8bf ahi = *(const v8bf*)&sA[bufi][wm + mt * 16 + r][16 + (half << 3)];
      v16bf a;
#pragma unroll
      for (int i = 0; i < 8; ++i) { a[i] = alo[i]; a[i + 8] = ahi[i]; }
#pragma unroll
      for (int nt = 0; nt < 4; ++nt)
        acc[mt][nt] = __builtin_amdgcn_wmma_f32_16x16x32_bf16(
            false, a, false, bfrag[nt], (short)0, acc[mt][nt], false, false);
    }
  };

#if USE_TDM
  const unsigned ldsA0 = (unsigned)(unsigned long long)(uintptr_t)&sA[0][0][0];
  const unsigned ldsA1 = (unsigned)(unsigned long long)(uintptr_t)&sA[1][0][0];
  const unsigned ldsB0 = (unsigned)(unsigned long long)(uintptr_t)&sB[0][0][0];
  const unsigned ldsB1 = (unsigned)(unsigned long long)(uintptr_t)&sB[1][0][0];
  const unsigned long long abase =
      (unsigned long long)(uintptr_t)(A + (long)m0 * Kk);
  const unsigned long long bbase =
      (unsigned long long)(uintptr_t)(Bw + (long)n0 * Kk);

  // prologue: DMA chunk 0 into buffer 0
  if (wid == 0) {
    tdm_load_tile_2d(abase, ldsA0, (unsigned)Kk, BM);
    tdm_load_tile_2d(bbase, ldsB0, (unsigned)Kk, BN);
    wait_tensorcnt0();
  }
  __syncthreads();

  int buf = 0;
  for (int k0 = 0; k0 < Kk; k0 += BK) {
    const bool more = (k0 + BK) < Kk;
    if (more && wid == 0) {           // DMA next chunk into the other buffer
      const unsigned long long off = (unsigned long long)(k0 + BK) * 2ull;
      tdm_load_tile_2d(abase + off, buf ? ldsA0 : ldsA1, (unsigned)Kk, BM);
      tdm_load_tile_2d(bbase + off, buf ? ldsB0 : ldsB1, (unsigned)Kk, BN);
    }
    compute_chunk(buf);               // WMMAs overlap the in-flight TDM
    if (more && wid == 0) wait_tensorcnt0();
    __syncthreads();
    buf ^= 1;
  }
#else
  // cooperative staging fallback: each thread moves 16 bf16 of A and of B
  const int srow = tid >> 1;
  const int sseg = (tid & 1) << 4;
  const bf16* Aptr = A  + (long)(m0 + srow) * Kk + sseg;
  const bf16* Bptr = Bw + (long)(n0 + srow) * Kk + sseg;
  for (int k0 = 0; k0 < Kk; k0 += BK) {
    *(v16bf*)&sA[0][srow][sseg] = *(const v16bf*)(Aptr + k0);
    *(v16bf*)&sB[0][srow][sseg] = *(const v16bf*)(Bptr + k0);
    if (k0 + 3 * BK <= Kk) {
      __builtin_prefetch(Aptr + k0 + 2 * BK, 0, 1);
      __builtin_prefetch(Bptr + k0 + 2 * BK, 0, 1);
    }
    __syncthreads();
    compute_chunk(0);
    __syncthreads();
  }
#endif

#pragma unroll
  for (int mt = 0; mt < 2; ++mt)
#pragma unroll
    for (int nt = 0; nt < 4; ++nt) {
      const int col = n0 + wn + nt * 16 + r;
      const float badd = bias ? bias[col] : 0.f;
      const int rowbase = m0 + wm + mt * 16 + (half << 3);
      v8f c = acc[mt][nt];
#pragma unroll
      for (int i = 0; i < 8; ++i) {
        const long idx = (long)(rowbase + i) * Nn + col;
        float v = c[i] + badd;
        if (resid) v += resid[idx];
        C[idx] = v;
      }
    }
}

// ---------------------------------------------------------------------------
// Elementwise / reduction kernels
// ---------------------------------------------------------------------------
__global__ void cvt_f32_bf16(const float* __restrict__ s, bf16* __restrict__ d, long n) {
  long i = (long)blockIdx.x * blockDim.x + threadIdx.x;
  if (i < n) d[i] = (bf16)s[i];
}

__global__ void cvt_pad_rows(const float* __restrict__ s, bf16* __restrict__ d,
                             int layers, int nsrc, int npad, int K) {
  long i = (long)blockIdx.x * blockDim.x + threadIdx.x;
  long total = (long)layers * npad * K;
  if (i >= total) return;
  int col = (int)(i % K);
  long rem = i / K;
  int row = (int)(rem % npad);
  int ly  = (int)(rem / npad);
  d[i] = (row < nsrc) ? (bf16)s[((long)ly * nsrc + row) * K + col] : (bf16)0.f;
}

__global__ void pad_bias_k(const float* __restrict__ s, float* __restrict__ d) {
  int i = blockIdx.x * blockDim.x + threadIdx.x;
  if (i < NH) d[i] = (i < NCLS) ? s[i] : 0.f;
}

__global__ void im2col_patch(const float* __restrict__ x, bf16* __restrict__ dst) {
  long i = (long)blockIdx.x * blockDim.x + threadIdx.x;
  if (i >= (long)MROWS * DM) return;
  int m = (int)(i / DM), k = (int)(i % DM);
  int b = m / TOKL, t = m % TOKL;
  int ph = t / 14, pw = t % 14;
  int c = k >> 8, rem = k & 255;
  int yy = rem >> 4, xx = rem & 15;
  dst[i] = (bf16)x[(((long)b * 3 + c) * 224 + ph * 16 + yy) * 224 + pw * 16 + xx];
}

__launch_bounds__(256)
__global__ void layernorm_to_bf16(const float* __restrict__ x,
                                  const float* __restrict__ g,
                                  const float* __restrict__ beta,
                                  bf16* __restrict__ out, int D) {
  const int row = blockIdx.x;
  const float* xr = x + (long)row * D;
  __shared__ float s1[256], s2[256];
  float a = 0.f, b = 0.f;
  for (int d = threadIdx.x; d < D; d += 256) { float v = xr[d]; a += v; b += v * v; }
  s1[threadIdx.x] = a; s2[threadIdx.x] = b;
  __syncthreads();
  for (int off = 128; off > 0; off >>= 1) {
    if ((int)threadIdx.x < off) {
      s1[threadIdx.x] += s1[threadIdx.x + off];
      s2[threadIdx.x] += s2[threadIdx.x + off];
    }
    __syncthreads();
  }
  float mu  = s1[0] / D;
  float var = s2[0] / D - mu * mu;
  float rs  = rsqrtf(var + 1e-5f);
  bf16* orow = out + (long)row * D;
  for (int d = threadIdx.x; d < D; d += 256)
    orow[d] = (bf16)((xr[d] - mu) * rs * g[d] + beta[d]);
}

__global__ void dwconv_silu(const float* __restrict__ xz, const float* __restrict__ cw,
                            const float* __restrict__ cb, float* __restrict__ xc,
                            bf16* __restrict__ xcb) {
  long i = (long)blockIdx.x * blockDim.x + threadIdx.x;
  if (i >= (long)MROWS * DI) return;
  int m = (int)(i / DI), d = (int)(i % DI);
  int b = m / TOKL, t = m % TOKL;
  float acc = cb[d];
#pragma unroll
  for (int j = 0; j < DCONV; ++j) {
    int tt = t + j - (DCONV - 1);
    if (tt >= 0) acc += xz[(long)(b * TOKL + tt) * DXZ + d] * cw[d * DCONV + j];
  }
  float s = acc * (1.f / (1.f + expf(-acc)));
  xc[i] = s;
  xcb[i] = (bf16)s;
}

__launch_bounds__(256)
__global__ void dt_softplus(const float* __restrict__ dbl, const float* __restrict__ w,
                            const float* __restrict__ bias, float* __restrict__ dt) {
  __shared__ float sd[DTR];
  const int m = blockIdx.y;
  const int d = blockIdx.x * 256 + threadIdx.x;
  if (threadIdx.x < DTR) sd[threadIdx.x] = dbl[(long)m * NXP + threadIdx.x];
  __syncthreads();
  float acc = bias[d];
#pragma unroll 7
  for (int rr = 0; rr < DTR; ++rr) acc += sd[rr] * w[d * DTR + rr];
  dt[(long)m * DI + d] = (acc > 20.f) ? acc : log1pf(expf(acc));
}

__launch_bounds__(256)
__global__ void sel_scan(const float* __restrict__ dbl, const float* __restrict__ dt,
                         const float* __restrict__ xc, const float* __restrict__ Alog,
                         float* __restrict__ ys) {
  const int b = blockIdx.y;
  const int d = blockIdx.x * 256 + threadIdx.x;
  float a[DST], hst[DST];
#pragma unroll
  for (int s = 0; s < DST; ++s) { a[s] = -expf(Alog[d * DST + s]); hst[s] = 0.f; }
  __shared__ float sB[DST], sC[DST];
  for (int t = 0; t < TOKL; ++t) {
    const int m = b * TOKL + t;
    if (threadIdx.x < DST)
      sB[threadIdx.x] = dbl[(long)m * NXP + DTR + threadIdx.x];
    else if (threadIdx.x < 2 * DST)
      sC[threadIdx.x - DST] = dbl[(long)m * NXP + DTR + threadIdx.x];
    __syncthreads();
    const float dtv = dt[(long)m * DI + d];
    const float xv  = xc[(long)m * DI + d];
    float y = 0.f;
#pragma unroll
    for (int s = 0; s < DST; ++s) {
      float dA = expf(dtv * a[s]);
      hst[s] = dA * hst[s] + dtv * sB[s] * xv;
      y += hst[s] * sC[s];
    }
    ys[(long)m * DI + d] = y;
    __syncthreads();
  }
}

__global__ void gate_silu(const float* __restrict__ ys, const float* __restrict__ xc,
                          const float* __restrict__ xz, const float* __restrict__ Dp,
                          bf16* __restrict__ yb) {
  long i = (long)blockIdx.x * blockDim.x + threadIdx.x;
  if (i >= (long)MROWS * DI) return;
  int m = (int)(i / DI), d = (int)(i % DI);
  float y = ys[i] + Dp[d] * xc[i];
  float z = xz[(long)m * DXZ + DI + d];
  float g = z * (1.f / (1.f + expf(-z)));
  yb[i] = (bf16)(y * g);
}

__launch_bounds__(256)
__global__ void pool_mean(const bf16* __restrict__ lnout, bf16* __restrict__ pooled) {
  const int d = blockIdx.x * 256 + threadIdx.x;   // < DM
  const int b = blockIdx.y;                        // < MH (zero-pad rows >= BATCH)
  float s = 0.f;
  if (b < BATCH) {
    for (int t = 0; t < TOKL; ++t) s += (float)lnout[(long)(b * TOKL + t) * DM + d];
    s *= (1.f / TOKL);
  }
  pooled[(long)b * DM + d] = (bf16)s;
}

__global__ void head_copy(const float* __restrict__ src, float* __restrict__ dst) {
  int i = blockIdx.x * blockDim.x + threadIdx.x;
  if (i >= BATCH * NCLS) return;
  int b = i / NCLS, c = i % NCLS;
  dst[i] = src[(long)b * NH + c];
}

// ---------------------------------------------------------------------------
// Launch
// ---------------------------------------------------------------------------
extern "C" void kernel_launch(void* const* d_in, const int* in_sizes, int n_in,
                              void* d_out, int out_size, void* d_ws, size_t ws_size,
                              hipStream_t stream) {
  (void)in_sizes; (void)n_in; (void)out_size; (void)ws_size;

  const float* x       = (const float*)d_in[0];
  const float* patch_w = (const float*)d_in[1];
  const float* patch_b = (const float*)d_in[2];
  const float* ln_g    = (const float*)d_in[3];
  const float* ln_b    = (const float*)d_in[4];
  const float* in_w    = (const float*)d_in[5];
  const float* conv_w  = (const float*)d_in[6];
  const float* conv_b  = (const float*)d_in[7];
  const float* xp_w    = (const float*)d_in[8];
  const float* dt_w    = (const float*)d_in[9];
  const float* dt_b    = (const float*)d_in[10];
  const float* A_log   = (const float*)d_in[11];
  const float* Dp      = (const float*)d_in[12];
  const float* out_w   = (const float*)d_in[13];
  const float* fn_g    = (const float*)d_in[14];
  const float* fn_b    = (const float*)d_in[15];
  const float* head_w  = (const float*)d_in[16];
  const float* head_b  = (const float*)d_in[17];
  float* out = (float*)d_out;

  char* base = (char*)d_ws;
  size_t off = 0;
  auto alloc = [&](size_t bytes) -> void* {
    void* p = base + off;
    off += (bytes + 255) & ~(size_t)255;
    return p;
  };
  float* h       = (float*)alloc((size_t)MROWS * DM * 4);
  bf16*  hb      = (bf16*) alloc((size_t)MROWS * DM * 2);
  float* xz      = (float*)alloc((size_t)MROWS * DXZ * 4);
  float* xc      = (float*)alloc((size_t)MROWS * DI * 4);
  bf16*  xcb     = (bf16*) alloc((size_t)MROWS * DI * 2);
  float* dbl     = (float*)alloc((size_t)MROWS * NXP * 4);
  float* dtb     = (float*)alloc((size_t)MROWS * DI * 4);
  float* ys      = (float*)alloc((size_t)MROWS * DI * 4);
  bf16*  yb      = (bf16*) alloc((size_t)MROWS * DI * 2);
  bf16*  pe      = (bf16*) alloc((size_t)MROWS * DM * 2);
  bf16*  pooled  = (bf16*) alloc((size_t)MH * DM * 2);
  float* headout = (float*)alloc((size_t)MH * NH * 4);
  float* headbp  = (float*)alloc((size_t)NH * 4);
  bf16*  w_patch = (bf16*) alloc((size_t)DM * DM * 2);
  bf16*  w_in    = (bf16*) alloc((size_t)DEPTH * DXZ * DM * 2);
  bf16*  w_xp    = (bf16*) alloc((size_t)DEPTH * NXP * DI * 2);
  bf16*  w_out   = (bf16*) alloc((size_t)DEPTH * DM * DI * 2);
  bf16*  w_head  = (bf16*) alloc((size_t)NH * DM * 2);

  auto g1 = [](long n) { return dim3((unsigned)((n + 255) / 256)); };

  // ---- weight preparation (fp32 -> bf16, padded where needed) ----
  cvt_f32_bf16<<<g1((long)DM * DM), 256, 0, stream>>>(patch_w, w_patch, (long)DM * DM);
  cvt_f32_bf16<<<g1((long)DEPTH * DXZ * DM), 256, 0, stream>>>(in_w, w_in, (long)DEPTH * DXZ * DM);
  cvt_f32_bf16<<<g1((long)DEPTH * DM * DI), 256, 0, stream>>>(out_w, w_out, (long)DEPTH * DM * DI);
  cvt_pad_rows<<<g1((long)DEPTH * NXP * DI), 256, 0, stream>>>(xp_w, w_xp, DEPTH, NXPRAW, NXP, DI);
  cvt_pad_rows<<<g1((long)NH * DM), 256, 0, stream>>>(head_w, w_head, 1, NCLS, NH, DM);
  pad_bias_k<<<g1(NH), 256, 0, stream>>>(head_b, headbp);

  // ---- patch embedding as GEMM (im2col K = 3*16*16 = 768) ----
  im2col_patch<<<g1((long)MROWS * DM), 256, 0, stream>>>(x, pe);
  gemm_bf16_wmma<<<dim3(DM / 128, MROWS / 128), 256, 0, stream>>>(
      pe, w_patch, patch_b, nullptr, h, MROWS, DM, DM);

  // ---- 12 Mamba blocks ----
  for (int i = 0; i < DEPTH; ++i) {
    layernorm_to_bf16<<<MROWS, 256, 0, stream>>>(h, ln_g + (size_t)i * DM,
                                                 ln_b + (size_t)i * DM, hb, DM);
    gemm_bf16_wmma<<<dim3(DXZ / 128, MROWS / 128), 256, 0, stream>>>(
        hb, w_in + (size_t)i * DXZ * DM, nullptr, nullptr, xz, MROWS, DXZ, DM);
    dwconv_silu<<<g1((long)MROWS * DI), 256, 0, stream>>>(
        xz, conv_w + (size_t)i * DI * DCONV, conv_b + (size_t)i * DI, xc, xcb);
    gemm_bf16_wmma<<<dim3(NXP / 128, MROWS / 128), 256, 0, stream>>>(
        xcb, w_xp + (size_t)i * NXP * DI, nullptr, nullptr, dbl, MROWS, NXP, DI);
    dt_softplus<<<dim3(DI / 256, MROWS), 256, 0, stream>>>(
        dbl, dt_w + (size_t)i * DI * DTR, dt_b + (size_t)i * DI, dtb);
    sel_scan<<<dim3(DI / 256, BATCH), 256, 0, stream>>>(
        dbl, dtb, xc, A_log + (size_t)i * DI * DST, ys);
    gate_silu<<<g1((long)MROWS * DI), 256, 0, stream>>>(
        ys, xc, xz, Dp + (size_t)i * DI, yb);
    gemm_bf16_wmma<<<dim3(DM / 128, MROWS / 128), 256, 0, stream>>>(
        yb, w_out + (size_t)i * DM * DI, nullptr, h, h, MROWS, DM, DI);
  }

  // ---- final LN -> mean pool -> head ----
  layernorm_to_bf16<<<MROWS, 256, 0, stream>>>(h, fn_g, fn_b, hb, DM);
  pool_mean<<<dim3(DM / 256, MH), 256, 0, stream>>>(hb, pooled);
  gemm_bf16_wmma<<<dim3(NH / 128, MH / 128), 256, 0, stream>>>(
      pooled, w_head, headbp, nullptr, headout, MH, NH, DM);
  head_copy<<<g1((long)BATCH * NCLS), 256, 0, stream>>>(headout, out);
}
